// AIVFIARForge_17489106829972
// MI455X (gfx1250) — matrix operand order repaired and checked by reading, not compiled, not run on hardware
//
#include <hip/hip_runtime.h>
#include <hip/hip_bf16.h>
#include <math.h>

// ---------------- problem constants (match reference) ----------------
constexpr int BV = 8192;   // batch
constexpr int DV = 1024;   // feature dim
constexpr int HV = 512;    // hidden (D/2)
constexpr int LV = 5;      // layers
constexpr size_t BD = (size_t)BV * DV;

typedef __attribute__((ext_vector_type(16))) __bf16        v16bf;
typedef __attribute__((ext_vector_type(8)))  float         v8f;
typedef __attribute__((ext_vector_type(8)))  unsigned short v8u;

union Frag {
  struct { v8u lo, hi; } h;
  v16bf v;
};

static __device__ __forceinline__ unsigned short f2bf(float f) {
  union { float f; unsigned int u; } c; c.f = f;
  unsigned int u = c.u;
  if ((u & 0x7F800000u) == 0x7F800000u) return (unsigned short)(u >> 16); // inf/nan
  return (unsigned short)((u + 0x7FFFu + ((u >> 16) & 1u)) >> 16);        // RNE
}

static __device__ __forceinline__ float sigmoidf_(float x) {
  return 1.0f / (1.0f + __expf(-x));
}

// =====================================================================
// Tiled bf16 GEMM: Y[M,N] = X[M,K] @ W[K,N] (+ bias), W given TRANSPOSED
// as Wt[N,K] so both A and B tiles are K-contiguous (no LDS transpose).
// Block: 256 threads (8 waves), tile 256(M) x 64(N), K step 32,
// double-buffered LDS so global loads overlap WMMA.
// Per wave: 2 M-tiles x 4 N-tiles -> 8 v_wmma_f32_16x16x32_bf16 / K-step.
// EPI: 0 = store f32 (pre-LN)   -> Yf
//      1 = tanh epilogue        -> patb (bf16), stabb (bf16 = next cur),
//                                  sum_st/sum_sq accumulation (variance)
//      2 = sigmoid epilogue     -> Yf (coherence field)
// =====================================================================
template<int EPI>
__global__ void __launch_bounds__(256)
gemm_bf16(const unsigned short* __restrict__ X,   // [M,K] bf16
          const unsigned short* __restrict__ Wt,  // [N,K] bf16 (transposed weights)
          const float* __restrict__ bias,         // [N] or nullptr
          int M, int N, int K,
          float* __restrict__ Yf,
          unsigned short* __restrict__ patb,
          unsigned short* __restrict__ stabb,
          float* __restrict__ sum_st,
          float* __restrict__ sum_sq,
          const float* __restrict__ sc,           // stability controller row [N]
          int accumulate)
{
  __shared__ alignas(16) unsigned short lA[2][256][40]; // [m][k] (+pad, 16B rows)
  __shared__ alignas(16) unsigned short lB[2][64][40];  // [n][k] (K-contiguous)

  const int tid  = threadIdx.x;
  const int lane = tid & 31;
  const int wv   = tid >> 5;            // 0..7
  const int m0   = blockIdx.y * 256;
  const int n0   = blockIdx.x * 64;

  v8f zero = {};
  v8f acc[2][4];
  #pragma unroll
  for (int t = 0; t < 2; ++t)
    #pragma unroll
    for (int j = 0; j < 4; ++j) acc[t][j] = zero;

  // staging: each thread owns 4 A-chunks (rows rS, rS+64, rS+128, rS+192) and
  // 1 B-chunk; all 16B, K-contiguous.
  const int rS = tid >> 2;              // 0..63
  const int cS = (tid & 3) * 8;         // k-chunk offset
  const unsigned short* pA = X  + (size_t)(m0 + rS) * K + cS;
  const unsigned short* pB = Wt + (size_t)(n0 + rS) * K + cS;

  // WMMA fragment addressing (ISA 16-bit A 16x32 / B 32x16 lane layouts)
  const int amBase = 32 * wv + (lane & 15);  // wave owns rows [32wv,32wv+32)
  const int akb = (lane >> 4) * 8;           // A: K base
  const int bkb = (lane >> 4) * 8;           // B: K base

  // preload tile 0 into registers
  v8u ra0 = *(const v8u*)(pA);
  v8u ra1 = *(const v8u*)(pA + (size_t)64  * K);
  v8u ra2 = *(const v8u*)(pA + (size_t)128 * K);
  v8u ra3 = *(const v8u*)(pA + (size_t)192 * K);
  v8u rb0 = *(const v8u*)(pB);

  int buf = 0;
  for (int k0 = 0; k0 < K; k0 += 32) {
    // ---- commit staged registers to LDS buffer `buf` ----
    *(v8u*)&lA[buf][rS      ][cS] = ra0;
    *(v8u*)&lA[buf][rS + 64 ][cS] = ra1;
    *(v8u*)&lA[buf][rS + 128][cS] = ra2;
    *(v8u*)&lA[buf][rS + 192][cS] = ra3;
    *(v8u*)&lB[buf][rS      ][cS] = rb0;
    __syncthreads();  // compiler emits s_wait_dscnt + barrier

    // ---- issue next tile's global loads (overlap with WMMA below) ----
    if (k0 + 32 < K) {
      const int kn = k0 + 32;
      ra0 = *(const v8u*)(pA + kn);
      ra1 = *(const v8u*)(pA + (size_t)64  * K + kn);
      ra2 = *(const v8u*)(pA + (size_t)128 * K + kn);
      ra3 = *(const v8u*)(pA + (size_t)192 * K + kn);
      rb0 = *(const v8u*)(pB + kn);
    }

    // ---- fragments + 8 WMMAs on current buffer ----
    Frag af0, af1;
    af0.h.lo = *(const v8u*)&lA[buf][amBase     ][akb];
    af0.h.hi = *(const v8u*)&lA[buf][amBase     ][akb + 16];
    af1.h.lo = *(const v8u*)&lA[buf][amBase + 16][akb];
    af1.h.hi = *(const v8u*)&lA[buf][amBase + 16][akb + 16];
    #pragma unroll
    for (int j = 0; j < 4; ++j) {
      const int bn = 16 * j + (lane & 15);
      Frag bf;
      bf.h.lo = *(const v8u*)&lB[buf][bn][bkb];
      bf.h.hi = *(const v8u*)&lB[buf][bn][bkb + 16];
      acc[0][j] = __builtin_amdgcn_wmma_f32_16x16x32_bf16(
          false, af0.v, false, bf.v, (short)0, acc[0][j], false, false);
      acc[1][j] = __builtin_amdgcn_wmma_f32_16x16x32_bf16(
          false, af1.v, false, bf.v, (short)0, acc[1][j], false, false);
    }
    buf ^= 1;
  }

  // ---- epilogue: C/D layout -> VGPR i is row i (+8 for upper lanes), col lane&15
  const int ccol = lane & 15;
  #pragma unroll
  for (int mt = 0; mt < 2; ++mt) {
    const int crow0 = m0 + 32 * wv + mt * 16 + ((lane >> 4) << 3);
    #pragma unroll
    for (int j = 0; j < 4; ++j) {
      const int col = n0 + 16 * j + ccol;
      const float bcol = bias ? bias[col] : 0.0f;
      float sstab = 0.0f;
      if constexpr (EPI == 1) sstab = sigmoidf_(sc[col]);
      #pragma unroll
      for (int i = 0; i < 8; ++i) {
        const size_t o = (size_t)(crow0 + i) * N + col;
        float y = acc[mt][j][i] + bcol;
        if constexpr (EPI == 0) {
          Yf[o] = y;
        } else if constexpr (EPI == 1) {
          float p  = tanhf(y);
          float st = p * sstab;
          patb[o]  = f2bf(p);
          stabb[o] = f2bf(st);
          if (accumulate) { sum_st[o] += st; sum_sq[o] += st * st; }
          else            { sum_st[o]  = st; sum_sq[o]  = st * st; }
        } else {
          Yf[o] = sigmoidf_(y);
        }
      }
    }
  }
}

// =====================================================================
// Row LayerNorm + activation. One block (256 threads) per row of length N.
// ACT: 0 = GELU(exact, erf) -> bf16 out
//      1 = sigmoid -> f32 accumulation (emo mean)
//      2 = ReLU -> bf16 out
// =====================================================================
template<int NPT, int ACT>
__global__ void __launch_bounds__(256)
ln_act(const float* __restrict__ Y,
       const float* __restrict__ g, const float* __restrict__ be,
       int N, unsigned short* __restrict__ outb,
       float* __restrict__ outacc, int accumulate)
{
  const int row = blockIdx.x;
  const int tid = threadIdx.x;
  __shared__ float rbuf[256];
  __shared__ float smean, srstd;

  float v[NPT];
  float s = 0.0f;
  #pragma unroll
  for (int i = 0; i < NPT; ++i) {
    v[i] = Y[(size_t)row * N + tid + i * 256];
    s += v[i];
  }
  rbuf[tid] = s; __syncthreads();
  for (int off = 128; off > 0; off >>= 1) {
    if (tid < off) rbuf[tid] += rbuf[tid + off];
    __syncthreads();
  }
  if (tid == 0) smean = rbuf[0] / (float)N;
  __syncthreads();
  const float mean = smean;

  float s2 = 0.0f;
  #pragma unroll
  for (int i = 0; i < NPT; ++i) { float d = v[i] - mean; s2 += d * d; }
  rbuf[tid] = s2; __syncthreads();
  for (int off = 128; off > 0; off >>= 1) {
    if (tid < off) rbuf[tid] += rbuf[tid + off];
    __syncthreads();
  }
  if (tid == 0) srstd = rsqrtf(rbuf[0] / (float)N + 1e-5f);
  __syncthreads();
  const float rstd = srstd;

  #pragma unroll
  for (int i = 0; i < NPT; ++i) {
    const int col = tid + i * 256;
    const size_t o = (size_t)row * N + col;
    float x = (v[i] - mean) * rstd * g[col] + be[col];
    if constexpr (ACT == 0) {
      outb[o] = f2bf(0.5f * x * (1.0f + erff(x * 0.70710678118654752f)));
    } else if constexpr (ACT == 1) {
      float e = sigmoidf_(x);
      if (accumulate) outacc[o] += e; else outacc[o] = e;
    } else {
      outb[o] = f2bf(fmaxf(x, 0.0f));
    }
  }
}

// ---------------- small helper kernels ----------------
__global__ void f32_to_bf16_k(const float* __restrict__ src,
                              unsigned short* __restrict__ dst, size_t n) {
  size_t i = (size_t)blockIdx.x * 256 + threadIdx.x;
  if (i < n) dst[i] = f2bf(src[i]);
}

// W[l][K][N] f32 -> Wt[l][N][K] bf16 (32x32 LDS tile transpose; both coalesced)
__global__ void __launch_bounds__(256)
transpose_cvt_k(const float* __restrict__ src, unsigned short* __restrict__ dst,
                int K, int N) {
  __shared__ float t[32][33];
  src += (size_t)blockIdx.z * K * N;
  dst += (size_t)blockIdx.z * K * N;
  const int bx = blockIdx.x * 32;   // N
  const int by = blockIdx.y * 32;   // K
  const int tx = threadIdx.x & 31, ty = threadIdx.x >> 5;  // 32x8
  #pragma unroll
  for (int i = 0; i < 32; i += 8)
    t[ty + i][tx] = src[(size_t)(by + ty + i) * N + bx + tx];
  __syncthreads();
  #pragma unroll
  for (int i = 0; i < 32; i += 8)
    dst[(size_t)(bx + ty + i) * K + by + tx] = f2bf(t[tx][ty + i]);
}

// transposed circulant: Ct[n][k] = C[k][n] = 0.9^((n-k) mod D)
__global__ void build_cmat_k(unsigned short* __restrict__ cmat) {
  int i = blockIdx.x * 256 + threadIdx.x;   // D*D elements
  int n = i >> 10, k = i & 1023;
  int idx = (n - k) & 1023;
  cmat[i] = f2bf(__powf(0.9f, (float)idx));
}

__global__ void zero_k(float* p) { p[0] = 0.0f; }

__global__ void finalize_pe(const float* __restrict__ sum_st,
                            const float* __restrict__ sum_sq,
                            const float* __restrict__ emo_sum,
                            float* __restrict__ out_comb,
                            float* __restrict__ out_emo,
                            unsigned short* __restrict__ comb_bf,
                            float* __restrict__ var_acc) {
  const int i = blockIdx.x * 256 + threadIdx.x;
  const float sm   = sum_st[i];
  const float mean = sm * (1.0f / (float)LV);
  out_comb[i] = mean;
  comb_bf[i]  = f2bf(mean);
  out_emo[i]  = emo_sum[i] * (1.0f / (float)LV);
  float lvar = (sum_sq[i] - sm * mean) * (1.0f / (float)(LV - 1));  // unbiased
  __shared__ float rbuf[256];
  rbuf[threadIdx.x] = lvar; __syncthreads();
  for (int off = 128; off > 0; off >>= 1) {
    if (threadIdx.x < off) rbuf[threadIdx.x] += rbuf[threadIdx.x + off];
    __syncthreads();
  }
  if (threadIdx.x == 0) atomicAdd(var_acc, rbuf[0]);
}

__global__ void stab_final_k(const float* __restrict__ var_acc,
                             float* __restrict__ out) {
  out[0] = 1.0f - var_acc[0] / (float)BD;
}

// =====================================================================
extern "C" void kernel_launch(void* const* d_in, const int* in_sizes, int n_in,
                              void* d_out, int out_size, void* d_ws, size_t ws_size,
                              hipStream_t stream)
{
  // ---- inputs (setup_inputs dict order) ----
  const float* seed   = (const float*)d_in[0];
  const float* pg_w1  = (const float*)d_in[2];
  const float* pg_b1  = (const float*)d_in[3];
  const float* pg_g1  = (const float*)d_in[4];
  const float* pg_be1 = (const float*)d_in[5];
  const float* pg_w2  = (const float*)d_in[6];
  const float* pg_b2  = (const float*)d_in[7];
  const float* es_w   = (const float*)d_in[8];
  const float* es_b   = (const float*)d_in[9];
  const float* es_g   = (const float*)d_in[10];
  const float* es_be  = (const float*)d_in[11];
  const float* qg_w1  = (const float*)d_in[12];
  const float* qg_b1  = (const float*)d_in[13];
  const float* qg_g1  = (const float*)d_in[14];
  const float* qg_be1 = (const float*)d_in[15];
  const float* qg_w2  = (const float*)d_in[16];
  const float* qg_b2  = (const float*)d_in[17];
  const float* stab_c = (const float*)d_in[18];
  float* out = (float*)d_out;

  // ---- workspace carve-up ----
  char* ws = (char*)d_ws;
  auto carve = [&](size_t bytes) -> char* {
    char* p = ws; ws += (bytes + 255) & ~(size_t)255; return p;
  };
  const size_t nPG1 = (size_t)LV * DV * 2 * DV;
  const size_t nPG2 = nPG1;
  const size_t nES  = (size_t)LV * DV * DV;
  const size_t nQ1  = (size_t)LV * DV * HV;
  const size_t nQ2  = (size_t)LV * HV * DV;

  unsigned short* wPG1 = (unsigned short*)carve(nPG1 * 2);  // [l][2D][D] (N,K)
  unsigned short* wPG2 = (unsigned short*)carve(nPG2 * 2);  // [l][D][2D]
  unsigned short* wES  = (unsigned short*)carve(nES * 2);   // [l][D][D]
  unsigned short* wQ1  = (unsigned short*)carve(nQ1 * 2);   // [l][H][D]
  unsigned short* wQ2  = (unsigned short*)carve(nQ2 * 2);   // [l][D][H]
  unsigned short* cur_bf  = (unsigned short*)carve(BD * 2);
  unsigned short* h1_bf   = (unsigned short*)carve(BD * 2 * 2);
  unsigned short* pat_bf  = (unsigned short*)carve(BD * 2);
  unsigned short* q1_bf   = (unsigned short*)carve((size_t)BV * HV * 2);
  unsigned short* comb_bf = (unsigned short*)carve(BD * 2);
  unsigned short* cmat_bf = (unsigned short*)carve((size_t)DV * DV * 2);
  float* ysc     = (float*)carve(BD * 2 * 4);
  float* sum_st  = (float*)carve(BD * 4);
  float* sum_sq  = (float*)carve(BD * 4);
  float* emo_sum = (float*)carve(BD * 4);
  float* var_acc = (float*)carve(256);

  // ---- init / conversions (weights transposed to [N][K] bf16) ----
  zero_k<<<1, 1, 0, stream>>>(var_acc);
  f32_to_bf16_k<<<dim3((unsigned)((BD + 255) / 256)), 256, 0, stream>>>(
      seed, cur_bf, BD);
  auto tcvt = [&](const float* s, unsigned short* d, int K, int N) {
    transpose_cvt_k<<<dim3(N / 32, K / 32, LV), 256, 0, stream>>>(s, d, K, N);
  };
  tcvt(pg_w1, wPG1, DV, 2 * DV);
  tcvt(pg_w2, wPG2, 2 * DV, DV);
  tcvt(es_w,  wES,  DV, DV);
  tcvt(qg_w1, wQ1,  DV, HV);
  tcvt(qg_w2, wQ2,  HV, DV);
  build_cmat_k<<<dim3((DV * DV) / 256), 256, 0, stream>>>(cmat_bf);

  // ---- layer loop ----
  for (int l = 0; l < LV; ++l) {
    const int acc = (l > 0) ? 1 : 0;
    // pg1: [B,D]x[D,2D]+b -> ysc
    gemm_bf16<0><<<dim3((2 * DV) / 64, BV / 256), 256, 0, stream>>>(
        cur_bf, wPG1 + (size_t)l * DV * 2 * DV, pg_b1 + (size_t)l * 2 * DV,
        BV, 2 * DV, DV, ysc, nullptr, nullptr, nullptr, nullptr, nullptr, 0);
    // LN + GELU -> h1_bf
    ln_act<8, 0><<<BV, 256, 0, stream>>>(
        ysc, pg_g1 + (size_t)l * 2 * DV, pg_be1 + (size_t)l * 2 * DV,
        2 * DV, h1_bf, nullptr, 0);
    // pg2: [B,2D]x[2D,D]+b -> tanh epilogue (pat, stable, Σst, Σst²)
    gemm_bf16<1><<<dim3(DV / 64, BV / 256), 256, 0, stream>>>(
        h1_bf, wPG2 + (size_t)l * 2 * DV * DV, pg_b2 + (size_t)l * DV,
        BV, DV, 2 * DV, nullptr, pat_bf, cur_bf, sum_st, sum_sq,
        stab_c + (size_t)l * DV, acc);
    // es: [B,D]x[D,D]+b -> ysc ; LN + sigmoid accumulate
    gemm_bf16<0><<<dim3(DV / 64, BV / 256), 256, 0, stream>>>(
        pat_bf, wES + (size_t)l * DV * DV, es_b + (size_t)l * DV,
        BV, DV, DV, ysc, nullptr, nullptr, nullptr, nullptr, nullptr, 0);
    ln_act<4, 1><<<BV, 256, 0, stream>>>(
        ysc, es_g + (size_t)l * DV, es_be + (size_t)l * DV,
        DV, nullptr, emo_sum, acc);
    // qg1: [B,D]x[D,H]+b -> ysc ; LN + ReLU -> q1_bf
    gemm_bf16<0><<<dim3(HV / 64, BV / 256), 256, 0, stream>>>(
        pat_bf, wQ1 + (size_t)l * DV * HV, qg_b1 + (size_t)l * HV,
        BV, HV, DV, ysc, nullptr, nullptr, nullptr, nullptr, nullptr, 0);
    ln_act<2, 2><<<BV, 256, 0, stream>>>(
        ysc, qg_g1 + (size_t)l * HV, qg_be1 + (size_t)l * HV,
        HV, q1_bf, nullptr, 0);
    // qg2: [B,H]x[H,D]+b -> anchors slice of d_out (f32)
    gemm_bf16<0><<<dim3(DV / 64, BV / 256), 256, 0, stream>>>(
        q1_bf, wQ2 + (size_t)l * HV * DV, qg_b2 + (size_t)l * DV,
        BV, DV, HV, out + 3 * BD + 1 + (size_t)l * BD,
        nullptr, nullptr, nullptr, nullptr, nullptr, 0);
  }

  // ---- finalize: means, variance partials, coherence, scalar ----
  finalize_pe<<<dim3((unsigned)(BD / 256)), 256, 0, stream>>>(
      sum_st, sum_sq, emo_sum, out, out + BD, comb_bf, var_acc);
  gemm_bf16<2><<<dim3(DV / 64, BV / 256), 256, 0, stream>>>(
      comb_bf, cmat_bf, nullptr, BV, DV, DV, out + 2 * BD,
      nullptr, nullptr, nullptr, nullptr, nullptr, 0);
  stab_final_k<<<1, 1, 0, stream>>>(var_acc, out + 3 * BD);
}
